// BlockWithFFN_40046275068187
// MI455X (gfx1250) — matrix operand order repaired and checked by reading, not compiled, not run on hardware
//
#include <hip/hip_runtime.h>

typedef __attribute__((ext_vector_type(16))) _Float16 v16h;
typedef __attribute__((ext_vector_type(8)))  float    v8f;

#define DD  6
#define SS  128
#define TPB 256
#define EPSF 1e-5f

// CDNA5 16-bit A-matrix 16x32 per-lane K map (ISA 7.12.2 table):
// element e (0..15), lane-half hi: k = (e>>3)*16 + hi*8 + (e&7)
__device__ __forceinline__ int akmap(int e, int hi) {
    return ((e >> 3) << 4) + hi * 8 + (e & 7);
}
// B-matrix 32x16: same K map as A (must match A's order):
// k = (e>>3)*16 + hi*8 + (e&7)
__device__ __forceinline__ int bkmap(int e, int hi) {
    return ((e >> 3) << 4) + hi * 8 + (e & 7);
}

__global__ __launch_bounds__(TPB, 1) void attn_block_kernel(
    const float* __restrict__ x,
    const float* __restrict__ ln1_w, const float* __restrict__ ln1_b,
    const float* __restrict__ wqkv,  const float* __restrict__ bqkv,
    const float* __restrict__ wo,    const float* __restrict__ bo,
    const float* __restrict__ ln2_w, const float* __restrict__ ln2_b,
    const float* __restrict__ w1,    const float* __restrict__ b1,
    const float* __restrict__ w2,    const float* __restrict__ b2,
    float* __restrict__ out)
{
    __shared__ float    xr[SS][DD];     // residual copy of x
    __shared__ _Float16 Qh[SS][32];     // zero-padded K-dim (6 -> 32)
    __shared__ _Float16 Kh[SS][32];
    __shared__ _Float16 Vh[SS][16];     // zero-padded N-dim (6 -> 16)
    __shared__ _Float16 Ph[SS][SS];     // softmax(P) in f16
    __shared__ float    Of[SS][DD];     // attention output

    const int tid = threadIdx.x;
    const int b   = blockIdx.x;
    const float* xb = x + (size_t)b * (SS * DD);

    // ---------------- load x tile (768 floats) ----------------
    for (int i = tid; i < SS * DD; i += TPB)
        ((float*)xr)[i] = xb[i];
    __syncthreads();

    // ---------------- LN1 + QKV projection (token per lane) ----------------
    if (tid < SS) {
        float xv[DD];
        float mu = 0.f;
        #pragma unroll
        for (int d = 0; d < DD; ++d) { xv[d] = xr[tid][d]; mu += xv[d]; }
        mu *= (1.0f / DD);
        float var = 0.f;
        #pragma unroll
        for (int d = 0; d < DD; ++d) { float u = xv[d] - mu; var += u * u; }
        var *= (1.0f / DD);
        float inv = rsqrtf(var + EPSF);
        float h[DD];
        #pragma unroll
        for (int d = 0; d < DD; ++d)
            h[d] = (xv[d] - mu) * inv * ln1_w[d] + ln1_b[d];

        #pragma unroll
        for (int j = 0; j < 3 * DD; ++j) {
            float a = bqkv[j];
            #pragma unroll
            for (int d = 0; d < DD; ++d) a += h[d] * wqkv[d * (3 * DD) + j];
            if (j < DD)          Qh[tid][j]          = (_Float16)a;
            else if (j < 2 * DD) Kh[tid][j - DD]     = (_Float16)a;
            else                 Vh[tid][j - 2 * DD] = (_Float16)a;
        }
        // zero the K-dim / N-dim padding so WMMA sees clean zeros
        #pragma unroll
        for (int k = DD; k < 32; ++k) { Qh[tid][k] = (_Float16)0.0f; Kh[tid][k] = (_Float16)0.0f; }
        #pragma unroll
        for (int n = DD; n < 16; ++n) Vh[tid][n] = (_Float16)0.0f;
    }
    __syncthreads();

    const int lane = tid & 31;
    const int wv   = tid >> 5;      // 8 waves, wave w owns rows 16w..16w+15
    const int hi   = lane >> 4;
    const int col  = lane & 15;
    const int m0   = wv * 16;

    // ---------------- scores = Q K^T / sqrt(D); causal softmax -------------
    v16h afrag;
    #pragma unroll
    for (int e = 0; e < 16; ++e)
        afrag[e] = Qh[m0 + col][akmap(e, hi)];

    v8f acc[8];
    #pragma unroll
    for (int t = 0; t < 8; ++t) {
        v16h bfrag;
        #pragma unroll
        for (int e = 0; e < 16; ++e)
            bfrag[e] = Kh[t * 16 + col][bkmap(e, hi)];
        v8f c = {};
        acc[t] = __builtin_amdgcn_wmma_f32_16x16x32_f16(
            false, afrag, false, bfrag, (short)0, c, false, false);
    }

    const float scale = 0.40824829046386302f;  // 1/sqrt(6)
    #pragma unroll
    for (int t = 0; t < 8; ++t) {
        #pragma unroll
        for (int r = 0; r < 8; ++r) {
            int mrow = m0 + r + 8 * hi;          // C layout: VGPR r -> M = r + 8*hi
            int ncol = t * 16 + col;             // lane -> N
            float s = acc[t][r] * scale;
            acc[t][r] = (ncol <= mrow) ? s : -1e30f;
        }
    }

    // softmax per row; each row lives in one 16-lane half -> xor masks 1..8
    #pragma unroll
    for (int r = 0; r < 8; ++r) {
        float mx = -1e30f;
        #pragma unroll
        for (int t = 0; t < 8; ++t) mx = fmaxf(mx, acc[t][r]);
        #pragma unroll
        for (int off = 8; off > 0; off >>= 1) mx = fmaxf(mx, __shfl_xor(mx, off, 32));
        float sum = 0.f;
        #pragma unroll
        for (int t = 0; t < 8; ++t) {
            float p = __expf(acc[t][r] - mx);    // v_exp_f32 fast path
            acc[t][r] = p;
            sum += p;
        }
        #pragma unroll
        for (int off = 8; off > 0; off >>= 1) sum += __shfl_xor(sum, off, 32);
        float is = 1.0f / sum;
        #pragma unroll
        for (int t = 0; t < 8; ++t) acc[t][r] *= is;
    }

    #pragma unroll
    for (int t = 0; t < 8; ++t)
        #pragma unroll
        for (int r = 0; r < 8; ++r)
            Ph[m0 + r + 8 * hi][t * 16 + col] = (_Float16)acc[t][r];
    __syncthreads();

    // ---------------- O = P @ V (K=128 as 4 chunks of 32) ------------------
    v8f oacc = {};
    #pragma unroll
    for (int kc = 0; kc < 4; ++kc) {
        v16h pa, vb;
        #pragma unroll
        for (int e = 0; e < 16; ++e) {
            pa[e] = Ph[m0 + col][kc * 32 + akmap(e, hi)];
            vb[e] = Vh[kc * 32 + bkmap(e, hi)][col];
        }
        oacc = __builtin_amdgcn_wmma_f32_16x16x32_f16(
            false, pa, false, vb, (short)0, oacc, false, false);
    }
    if (col < DD) {
        #pragma unroll
        for (int r = 0; r < 8; ++r)
            Of[m0 + r + 8 * hi][col] = oacc[r];
    }
    __syncthreads();

    // -------- out-proj + residual + LN2 + GELU FFN + residual --------------
    if (tid < SS) {
        float ov[DD], xn[DD];
        #pragma unroll
        for (int d = 0; d < DD; ++d) ov[d] = Of[tid][d];
        #pragma unroll
        for (int d = 0; d < DD; ++d) {
            float a = bo[d];
            #pragma unroll
            for (int e = 0; e < DD; ++e) a += ov[e] * wo[e * DD + d];
            xn[d] = xr[tid][d] + a;
        }
        // LN2
        float mu = 0.f;
        #pragma unroll
        for (int d = 0; d < DD; ++d) mu += xn[d];
        mu *= (1.0f / DD);
        float var = 0.f;
        #pragma unroll
        for (int d = 0; d < DD; ++d) { float u = xn[d] - mu; var += u * u; }
        var *= (1.0f / DD);
        float inv = rsqrtf(var + EPSF);
        float h2[DD];
        #pragma unroll
        for (int d = 0; d < DD; ++d)
            h2[d] = (xn[d] - mu) * inv * ln2_w[d] + ln2_b[d];
        // FFN: gelu(h2 @ w1 + b1) @ w2 + b2 (exact gelu)
        float g[DD];
        #pragma unroll
        for (int j = 0; j < DD; ++j) {
            float f = b1[j];
            #pragma unroll
            for (int d = 0; d < DD; ++d) f += h2[d] * w1[d * DD + j];
            g[j] = 0.5f * f * (1.0f + erff(f * 0.70710678118654752f));
        }
        float* ob = out + (size_t)b * (SS * DD) + tid * DD;
        #pragma unroll
        for (int d = 0; d < DD; ++d) {
            float f = b2[d];
            #pragma unroll
            for (int j = 0; j < DD; ++j) f += g[j] * w2[j * DD + d];
            ob[d] = xn[d] + f;
        }
    }
}

extern "C" void kernel_launch(void* const* d_in, const int* in_sizes, int n_in,
                              void* d_out, int out_size, void* d_ws, size_t ws_size,
                              hipStream_t stream) {
    const float* x     = (const float*)d_in[0];
    const float* ln1_w = (const float*)d_in[1];
    const float* ln1_b = (const float*)d_in[2];
    const float* wqkv  = (const float*)d_in[3];
    const float* bqkv  = (const float*)d_in[4];
    const float* wo    = (const float*)d_in[5];
    const float* bo    = (const float*)d_in[6];
    const float* ln2_w = (const float*)d_in[7];
    const float* ln2_b = (const float*)d_in[8];
    const float* w1    = (const float*)d_in[9];
    const float* b1    = (const float*)d_in[10];
    const float* w2    = (const float*)d_in[11];
    const float* b2    = (const float*)d_in[12];
    float* out = (float*)d_out;

    const int B = in_sizes[0] / (SS * DD);   // 8192 sequences
    attn_block_kernel<<<B, TPB, 0, stream>>>(
        x, ln1_w, ln1_b, wqkv, bqkv, wo, bo, ln2_w, ln2_b, w1, b1, w2, b2, out);
}
